// DecoderWithContext_16956530884705
// MI455X (gfx1250) — matrix operand order repaired and checked by reading, not compiled, not run on hardware
//
#include <hip/hip_runtime.h>
#include <hip/hip_bf16.h>
#include <math.h>

// ---------------------------------------------------------------------------
// Problem constants (from the reference)
// ---------------------------------------------------------------------------
#define Bb   64
#define P    196          // 14*14
#define Ee   2048
#define Ll   30
#define Tt   29           // L-1 steps
#define Vv   10000
#define EMBe 512
#define AMLe 768
#define ADLe 500
#define Dd   1012         // 512+500
#define Aa   1012

typedef __attribute__((ext_vector_type(16))) _Float16 v16h;
typedef __attribute__((ext_vector_type(8)))  float    v8f;
typedef __attribute__((ext_vector_type(4)))  _Float16 h4;
typedef __attribute__((ext_vector_type(2)))  _Float16 h2;

// ---------------------------------------------------------------------------
// Skinny-M WMMA GEMM:  C[M x N] = A1[M x K1] @ W1 (+ A2[M x K2] @ W2) + bias
//   block = 128 thr = 4 waves; wave w owns 16-row m-tile w of a 64-row group.
//   Workgroup computes a 64x64 output block: one A fragment + four distinct
//   B fragments feed 4 back-to-back v_wmma per 32-deep k-tile (no WAR chains,
//   partial dscnt waits). LDS ping-pong double buffered (one barrier/k-tile).
//   Staging guards are uniform scalar branches; fast path issues all global
//   b128 loads back-to-back. f32 -> f16 at staging, f32 accumulate in WMMA.
// ---------------------------------------------------------------------------
template<bool HASA2, bool HASRMAP, bool HASMASK>
__global__ __launch_bounds__(128) void gemm64_wmma(
    const float* __restrict__ A1, int K1,
    const float* __restrict__ A2, int K2,
    const float* __restrict__ W1, const float* __restrict__ W2,
    const float* __restrict__ bias,
    float* __restrict__ C, long ldc, int N,
    const int* __restrict__ rmap, int group,
    const int* __restrict__ dlens, int step)
{
    __shared__ _Float16 As[2][64 * 36];   // [row][k], padded stride 36 halves
    __shared__ _Float16 Bs[2][64 * 36];   // [n][k] transposed, 64 cols

    const int tid   = threadIdx.x;
    const int wave  = tid >> 5;
    const int lane  = tid & 31;
    const int row16 = lane & 15;
    const int khalf = lane >> 4;
    const int nbase = blockIdx.x * 64;
    const int mbase = blockIdx.y * 64;
    const int Ktot  = K1 + K2;
    const int ntile = (Ktot + 31) >> 5;
    const bool fullN = (nbase + 64 <= N);

    // --- per-thread staging geometry (constant across k-tiles) ---
    const int ar = tid >> 3;          // A rows ar+16i, i=0..3
    const int ka = (tid & 7) * 4;     // A k offset (float4 slot)
    const float* a1p[4];
    const float* a2p[4];
    #pragma unroll
    for (int i = 0; i < 4; ++i) {
        int grow = mbase + ar + 16 * i;
        int src  = grow;
        if (HASRMAP) src = rmap[grow / group] * group + grow % group;
        a1p[i] = A1 + (long)src * K1;
        a2p[i] = HASA2 ? (A2 + (long)grow * K2) : nullptr;
    }
    // B: 256 chunks of (4 cols x 2 k-rows); thread handles chunks tid, tid+128
    int bnb[2], bkb[2];
    #pragma unroll
    for (int j = 0; j < 2; ++j) {
        int cid = tid + j * 128;
        bnb[j] = cid & 15;          // col base = nbase + bnb*4
        bkb[j] = (cid >> 4) * 2;    // k base within tile
    }

    auto stage = [&](int buf, int k0) {
        // L2 prefetch of the tile after next
        if (k0 + 64 < Ktot) {
            int kn = k0 + 64;
            const float* pf = (!HASA2 || kn < K1)
                                  ? &W1[(long)kn * N + nbase]
                                  : &W2[(long)(kn - K1) * N + nbase];
            __builtin_prefetch(pf, 0, 1);
        }

        const bool fullK = (k0 + 32 <= Ktot);
        const bool seg1  = (!HASA2 || k0 + 32 <= K1);   // tile fully in pair 1
        const bool seg2  = (HASA2 && k0 >= K1);         // tile fully in pair 2

        // ---- gather global data (fast paths are uniform, loads cluster) ----
        float4 av[4];
        if (fullK && seg1) {
            #pragma unroll
            for (int i = 0; i < 4; ++i)
                av[i] = *(const float4*)(a1p[i] + k0 + ka);
        } else if (fullK && seg2) {
            #pragma unroll
            for (int i = 0; i < 4; ++i)
                av[i] = *(const float4*)(a2p[i] + (k0 - K1) + ka);
        } else {
            #pragma unroll
            for (int i = 0; i < 4; ++i) {
                int k = k0 + ka;
                float4 v = make_float4(0.f, 0.f, 0.f, 0.f);
                if (k < Ktot)
                    v = (!HASA2 || k < K1) ? *(const float4*)(a1p[i] + k)
                                           : *(const float4*)(a2p[i] + k - K1);
                av[i] = v;
            }
        }

        float4 bw0[2], bw1[2];
        if (fullN && fullK && seg1) {
            #pragma unroll
            for (int j = 0; j < 2; ++j) {
                long k   = k0 + bkb[j];
                int  col = nbase + bnb[j] * 4;
                bw0[j] = *(const float4*)(W1 + k * N + col);
                bw1[j] = *(const float4*)(W1 + (k + 1) * N + col);
            }
        } else if (fullN && fullK && seg2) {
            #pragma unroll
            for (int j = 0; j < 2; ++j) {
                long k   = (long)(k0 - K1) + bkb[j];
                int  col = nbase + bnb[j] * 4;
                bw0[j] = *(const float4*)(W2 + k * N + col);
                bw1[j] = *(const float4*)(W2 + (k + 1) * N + col);
            }
        } else {
            #pragma unroll
            for (int j = 0; j < 2; ++j) {
                int k   = k0 + bkb[j];
                int col = nbase + bnb[j] * 4;
                float4 w0 = make_float4(0.f, 0.f, 0.f, 0.f);
                float4 w1 = w0;
                if (col + 4 <= N && k < Ktot) {   // all N mult of 4, K even
                    if (!HASA2 || k < K1) {
                        w0 = *(const float4*)(W1 + (long)k * N + col);
                        w1 = *(const float4*)(W1 + (long)(k + 1) * N + col);
                    } else {
                        w0 = *(const float4*)(W2 + (long)(k - K1) * N + col);
                        w1 = *(const float4*)(W2 + (long)(k + 1 - K1) * N + col);
                    }
                }
                bw0[j] = w0; bw1[j] = w1;
            }
        }

        // ---- convert + LDS store (after all loads issued) ----
        #pragma unroll
        for (int i = 0; i < 4; ++i) {
            h4 hv;
            hv[0] = (_Float16)av[i].x; hv[1] = (_Float16)av[i].y;
            hv[2] = (_Float16)av[i].z; hv[3] = (_Float16)av[i].w;
            *(h4*)&As[buf][(ar + 16 * i) * 36 + ka] = hv;
        }
        #pragma unroll
        for (int j = 0; j < 2; ++j) {
            h2 p0; p0[0] = (_Float16)bw0[j].x; p0[1] = (_Float16)bw1[j].x;
            h2 p1; p1[0] = (_Float16)bw0[j].y; p1[1] = (_Float16)bw1[j].y;
            h2 p2; p2[0] = (_Float16)bw0[j].z; p2[1] = (_Float16)bw1[j].z;
            h2 p3; p3[0] = (_Float16)bw0[j].w; p3[1] = (_Float16)bw1[j].w;
            _Float16* bp = &Bs[buf][(bnb[j] * 4) * 36 + bkb[j]];
            *(h2*)(bp + 0 * 36) = p0;
            *(h2*)(bp + 1 * 36) = p1;
            *(h2*)(bp + 2 * 36) = p2;
            *(h2*)(bp + 3 * 36) = p3;
        }
    };

    v8f acc0 = {}, acc1 = {}, acc2 = {}, acc3 = {};

    stage(0, 0);
    for (int kt = 0; kt < ntile; ++kt) {
        __syncthreads();
        if (kt + 1 < ntile) stage((kt + 1) & 1, (kt + 1) * 32);
        const int buf = kt & 1;

        // A fragment: contiguous k = 16*khalf + j (matches B)
        v16h a;
        const _Float16* arow = &As[buf][(wave * 16 + row16) * 36];
        #pragma unroll
        for (int j = 0; j < 16; ++j)
            a[j] = arow[j + (khalf << 4)];

        // 4 B fragments into distinct registers, then 4 back-to-back WMMAs
        const _Float16* bb = &Bs[buf][row16 * 36 + (khalf << 4)];
        v16h b0, b1, b2, b3;
        #pragma unroll
        for (int j = 0; j < 16; ++j) b0[j] = bb[j];
        #pragma unroll
        for (int j = 0; j < 16; ++j) b1[j] = bb[16 * 36 + j];
        #pragma unroll
        for (int j = 0; j < 16; ++j) b2[j] = bb[32 * 36 + j];
        #pragma unroll
        for (int j = 0; j < 16; ++j) b3[j] = bb[48 * 36 + j];

        acc0 = __builtin_amdgcn_wmma_f32_16x16x32_f16(
                   false, a, false, b0, (short)0, acc0, false, false);
        acc1 = __builtin_amdgcn_wmma_f32_16x16x32_f16(
                   false, a, false, b1, (short)0, acc1, false, false);
        acc2 = __builtin_amdgcn_wmma_f32_16x16x32_f16(
                   false, a, false, b2, (short)0, acc2, false, false);
        acc3 = __builtin_amdgcn_wmma_f32_16x16x32_f16(
                   false, a, false, b3, (short)0, acc3, false, false);
    }

    // C/D layout: lane -> N = lane%16; vgpr v -> M = v + 8*(lane/16)
    #pragma unroll
    for (int u = 0; u < 4; ++u) {
        v8f accu = (u == 0) ? acc0 : (u == 1) ? acc1 : (u == 2) ? acc2 : acc3;
        int col = nbase + u * 16 + row16;
        if (col < N) {
            float bv = bias ? bias[col] : 0.0f;
            #pragma unroll
            for (int v = 0; v < 8; ++v) {
                int grow  = mbase + wave * 16 + v + (khalf << 3);
                float val = accu[v] + bv;
                if (HASMASK && dlens[grow] <= step) val = 0.0f;
                C[(long)grow * ldc + col] = val;
            }
        }
    }
}

// ---------------------------------------------------------------------------
// Stable descending argsort of caption lengths (B=64), sorted caps, dlens
// ---------------------------------------------------------------------------
__global__ void sort_kernel(const int* __restrict__ cap_len,
                            const int* __restrict__ caps,
                            int* __restrict__ sind, int* __restrict__ dlens,
                            int* __restrict__ caps_s,
                            float* __restrict__ out_caps,
                            float* __restrict__ out_dlens,
                            float* __restrict__ out_sind)
{
    int i  = threadIdx.x;      // 64 threads
    int li = cap_len[i];
    int rank = 0;
    for (int j = 0; j < Bb; ++j) {
        int lj = cap_len[j];
        if (lj > li || (lj == li && j < i)) rank++;
    }
    sind[rank] = i;
    __syncthreads();
    int s  = sind[i];
    int dl = cap_len[s] - 1;
    dlens[i]     = dl;
    out_dlens[i] = (float)dl;
    out_sind[i]  = (float)s;
    for (int l = 0; l < Ll; ++l) {
        int v = caps[s * Ll + l];
        caps_s[i * Ll + l]   = v;
        out_caps[i * Ll + l] = (float)v;
    }
}

// enc.mean(axis=pixels) into dec_in[:, 0:E]
__global__ void encmean_kernel(const float* __restrict__ enc,
                               const int* __restrict__ sind,
                               float* __restrict__ dec_in)
{
    int gid = blockIdx.x * 256 + threadIdx.x;   // 64*2048
    int b = gid >> 11, e = gid & 2047;
    long base = (long)sind[b] * P * Ee + e;
    float s = 0.0f;
    for (int p = 0; p < P; ++p) s += enc[base + (long)p * Ee];
    dec_in[b * (Ee + ADLe) + e] = s * (1.0f / (float)P);
}

// scores[b,p] = relu(att1[b,p,:] + att2[b,:]) . W_full + b_full  (wave per (b,p))
__global__ __launch_bounds__(256) void scores_kernel(
    const float* __restrict__ att1, const float* __restrict__ att2,
    const float* __restrict__ W_full, const float* __restrict__ b_full,
    float* __restrict__ scores)
{
    int w    = blockIdx.x * 8 + (threadIdx.x >> 5);
    int lane = threadIdx.x & 31;
    if (w >= Bb * P) return;
    int b = w / P;
    const float* a1 = att1 + (long)w * Aa;
    const float* a2 = att2 + (long)b * Aa;
    float sum = 0.0f;
    for (int a = lane; a < Aa; a += 32) {
        float v = a1[a] + a2[a];
        v = fmaxf(v, 0.0f);
        sum += v * W_full[a];
    }
    for (int off = 16; off > 0; off >>= 1) sum += __shfl_xor(sum, off, 32);
    if (lane == 0) scores[w] = sum + b_full[0];
}

// softmax over 196 pixels; also write masked alpha straight into d_out
__global__ __launch_bounds__(256) void softmax_kernel(
    const float* __restrict__ scores, float* __restrict__ alpha,
    float* __restrict__ out_alphas, const int* __restrict__ dlens, int t)
{
    __shared__ float red[256];
    __shared__ float s_max, s_sum;
    int b = blockIdx.x, tid = threadIdx.x;
    float v = (tid < P) ? scores[b * P + tid] : -3.4e38f;
    red[tid] = v; __syncthreads();
    for (int s = 128; s > 0; s >>= 1) {
        if (tid < s) red[tid] = fmaxf(red[tid], red[tid + s]);
        __syncthreads();
    }
    if (tid == 0) s_max = red[0];
    __syncthreads();
    float e = (tid < P) ? __expf(v - s_max) : 0.0f;
    red[tid] = e; __syncthreads();
    for (int s = 128; s > 0; s >>= 1) {
        if (tid < s) red[tid] += red[tid + s];
        __syncthreads();
    }
    if (tid == 0) s_sum = red[0];
    __syncthreads();
    if (tid < P) {
        float al = e / s_sum;
        alpha[b * P + tid] = al;
        bool active = dlens[b] > t;
        out_alphas[(long)b * (Tt * P) + (long)t * P + tid] = active ? al : 0.0f;
    }
}

// awe[b,e] = sum_p alpha[b,p] * enc[sind[b],p,e]
__global__ __launch_bounds__(256) void awe_kernel(
    const float* __restrict__ alpha, const float* __restrict__ enc,
    const int* __restrict__ sind, float* __restrict__ awe)
{
    __shared__ float sa[P];
    int b = blockIdx.x, tid = threadIdx.x;
    if (tid < P) sa[tid] = alpha[b * P + tid];
    __syncthreads();
    long base = (long)sind[b] * P * Ee;
    for (int e = tid; e < Ee; e += 256) {
        float s = 0.0f;
        for (int p = 0; p < P; ++p) s += sa[p] * enc[base + (long)p * Ee + e];
        awe[b * Ee + e] = s;
    }
}

// x = [ E_emb[caps_s[:,t]] , sigmoid(h@W_fb + b_fb) * awe ]
__global__ void xbuild_kernel(const int* __restrict__ caps_s, int t,
                              const float* __restrict__ E_emb,
                              const float* __restrict__ fbg,
                              const float* __restrict__ b_fb,
                              const float* __restrict__ awe,
                              float* __restrict__ x)
{
    int gid = blockIdx.x * 256 + threadIdx.x;   // 64*2560
    int b = gid / (EMBe + Ee), j = gid % (EMBe + Ee);
    float v;
    if (j < EMBe) {
        int tok = caps_s[b * Ll + t];
        v = E_emb[(long)tok * EMBe + j];
    } else {
        int e = j - EMBe;
        float g = fbg[b * Ee + e] + b_fb[e];
        g = 1.0f / (1.0f + __expf(-g));
        v = g * awe[b * Ee + e];
    }
    x[b * (EMBe + Ee) + j] = v;
}

// LSTM pointwise; masked in-place h/c update (active rows only)
__global__ void lstm_kernel(const float* __restrict__ gates,
                            const float* __restrict__ b_hh,
                            const int* __restrict__ dlens, int t,
                            float* __restrict__ h, float* __restrict__ c)
{
    int gid = blockIdx.x * 256 + threadIdx.x;   // 64*1012
    if (gid >= Bb * Dd) return;
    int b = gid / Dd, d = gid % Dd;
    const float* g = gates + (long)b * (4 * Dd);
    float gi = g[d]          + b_hh[d];
    float gf = g[Dd + d]     + b_hh[Dd + d];
    float gg = g[2 * Dd + d] + b_hh[2 * Dd + d];
    float go = g[3 * Dd + d] + b_hh[3 * Dd + d];
    float si = 1.0f / (1.0f + __expf(-gi));
    float sf = 1.0f / (1.0f + __expf(-gf));
    float so = 1.0f / (1.0f + __expf(-go));
    float cn = sf * c[gid] + si * tanhf(gg);
    float hn = so * tanhf(cn);
    if (dlens[b] > t) { c[gid] = cn; h[gid] = hn; }
}

// ---------------------------------------------------------------------------
// Host driver
// ---------------------------------------------------------------------------
extern "C" void kernel_launch(void* const* d_in, const int* in_sizes, int n_in,
                              void* d_out, int out_size, void* d_ws, size_t ws_size,
                              hipStream_t stream) {
    const float* encoder_out = (const float*)d_in[0];
    const int*   enc_caps    = (const int*)  d_in[1];
    const int*   cap_len     = (const int*)  d_in[2];
    const float* arts        = (const float*)d_in[3];
    const float* E_emb       = (const float*)d_in[6];
    const float* W_par = (const float*)d_in[7];
    const float* b_par = (const float*)d_in[8];
    const float* W_enc = (const float*)d_in[9];
    const float* b_enc = (const float*)d_in[10];
    const float* W_dec = (const float*)d_in[11];
    const float* b_dec = (const float*)d_in[12];
    const float* W_full= (const float*)d_in[13];
    const float* b_full= (const float*)d_in[14];
    const float* W_h   = (const float*)d_in[15];
    const float* b_h   = (const float*)d_in[16];
    const float* W_c   = (const float*)d_in[17];
    const float* b_c   = (const float*)d_in[18];
    const float* W_fb  = (const float*)d_in[19];
    const float* b_fb  = (const float*)d_in[20];
    const float* W_ih  = (const float*)d_in[21];
    const float* b_ih  = (const float*)d_in[22];
    const float* W_hh  = (const float*)d_in[23];
    const float* b_hh  = (const float*)d_in[24];
    const float* W_fc  = (const float*)d_in[25];
    const float* b_fc  = (const float*)d_in[26];

    // output layout (tuple flattened, all as float)
    float* out        = (float*)d_out;
    float* out_preds  = out;                                   // B*T*V
    float* out_caps   = out + (long)Bb * Tt * Vv;              // B*L
    float* out_dlens  = out_caps + Bb * Ll;                    // B
    float* out_alphas = out_dlens + Bb;                        // B*T*P
    float* out_sind   = out_alphas + (long)Bb * Tt * P;        // B

    // workspace carve
    char* w = (char*)d_ws;
    auto carve = [&](size_t bytes) -> void* {
        void* p = (void*)w;
        w += (bytes + 255) & ~(size_t)255;
        return p;
    };
    float* att1   = (float*)carve((size_t)Bb * P * Aa * 4);    // 12544 x 1012
    int*   sind   = (int*)  carve(Bb * 4);
    int*   dlens  = (int*)  carve(Bb * 4);
    int*   caps_s = (int*)  carve(Bb * Ll * 4);
    float* dec_in = (float*)carve((size_t)Bb * (Ee + ADLe) * 4);
    float* hbuf   = (float*)carve((size_t)Bb * Dd * 4);
    float* cbuf   = (float*)carve((size_t)Bb * Dd * 4);
    float* att2   = (float*)carve((size_t)Bb * Aa * 4);
    float* scores = (float*)carve((size_t)Bb * P * 4);
    float* alpha  = (float*)carve((size_t)Bb * P * 4);
    float* awe    = (float*)carve((size_t)Bb * Ee * 4);
    float* fbg    = (float*)carve((size_t)Bb * Ee * 4);
    float* xbuf   = (float*)carve((size_t)Bb * (EMBe + Ee) * 4);
    float* gates  = (float*)carve((size_t)Bb * 4 * Dd * 4);

    dim3 blk128(128), blk256(256);
    auto ng = [](int N) { return (unsigned)((N + 63) / 64); };

    // --- setup ---
    sort_kernel<<<1, 64, 0, stream>>>(cap_len, enc_caps, sind, dlens, caps_s,
                                      out_caps, out_dlens, out_sind);
    encmean_kernel<<<(Bb * Ee) / 256, blk256, 0, stream>>>(encoder_out, sind, dec_in);
    // context = arts@W_par + b_par -> dec_in[:, E:E+ADL]   (rows gathered by sind)
    gemm64_wmma<false, true, false><<<dim3(ng(ADLe), 1), blk128, 0, stream>>>(
        arts, AMLe, nullptr, 0, W_par, nullptr, b_par,
        dec_in + Ee, Ee + ADLe, ADLe, sind, 1, nullptr, 0);
    // h0 / c0
    gemm64_wmma<false, false, false><<<dim3(ng(Dd), 1), blk128, 0, stream>>>(
        dec_in, Ee + ADLe, nullptr, 0, W_h, nullptr, b_h,
        hbuf, Dd, Dd, nullptr, 1, nullptr, 0);
    gemm64_wmma<false, false, false><<<dim3(ng(Dd), 1), blk128, 0, stream>>>(
        dec_in, Ee + ADLe, nullptr, 0, W_c, nullptr, b_c,
        cbuf, Dd, Dd, nullptr, 1, nullptr, 0);
    // att1 = enc_sorted @ W_enc + b_enc   (12544 x 1012; rows gathered by sind)
    gemm64_wmma<false, true, false><<<dim3(ng(Aa), (Bb * P) / 64), blk128, 0, stream>>>(
        encoder_out, Ee, nullptr, 0, W_enc, nullptr, b_enc,
        att1, Aa, Aa, sind, P, nullptr, 0);

    // --- decode steps ---
    for (int t = 0; t < Tt; ++t) {
        // att2 = h @ W_dec + b_dec
        gemm64_wmma<false, false, false><<<dim3(ng(Aa), 1), blk128, 0, stream>>>(
            hbuf, Dd, nullptr, 0, W_dec, nullptr, b_dec,
            att2, Aa, Aa, nullptr, 1, nullptr, 0);
        scores_kernel<<<(Bb * P) / 8, blk256, 0, stream>>>(att1, att2, W_full, b_full, scores);
        softmax_kernel<<<Bb, blk256, 0, stream>>>(scores, alpha, out_alphas, dlens, t);
        awe_kernel<<<Bb, blk256, 0, stream>>>(alpha, encoder_out, sind, awe);
        // fbg = h @ W_fb (bias folded into xbuild with sigmoid)
        gemm64_wmma<false, false, false><<<dim3(ng(Ee), 1), blk128, 0, stream>>>(
            hbuf, Dd, nullptr, 0, W_fb, nullptr, nullptr,
            fbg, Ee, Ee, nullptr, 1, nullptr, 0);
        xbuild_kernel<<<(Bb * (EMBe + Ee)) / 256, blk256, 0, stream>>>(
            caps_s, t, E_emb, fbg, b_fb, awe, xbuf);
        // gates = x @ W_ih + h @ W_hh + b_ih  (b_hh added in lstm kernel)
        gemm64_wmma<true, false, false><<<dim3(ng(4 * Dd), 1), blk128, 0, stream>>>(
            xbuf, EMBe + Ee, hbuf, Dd, W_ih, W_hh, b_ih,
            gates, 4 * Dd, 4 * Dd, nullptr, 1, nullptr, 0);
        lstm_kernel<<<(Bb * Dd + 255) / 256, blk256, 0, stream>>>(
            gates, b_hh, dlens, t, hbuf, cbuf);
        // preds = h_new @ W_fc + b_fc, masked, written into d_out[b][t][:]
        gemm64_wmma<false, false, true><<<dim3(ng(Vv), 1), blk128, 0, stream>>>(
            hbuf, Dd, nullptr, 0, W_fc, nullptr, b_fc,
            out_preds + (long)t * Vv, (long)Tt * Vv, Vv, nullptr, 1, dlens, t);
    }
}
